// GA_Edgeconv_32298154066799
// MI455X (gfx1250) — matrix operand order repaired and checked by reading, not compiled, not run on hardware
//
#include <hip/hip_runtime.h>
#include <stdint.h>

#define BATCH 8
#define NPTS  4096
#define CIN   64
#define COUT  64
#define KNN   16
#define GEOC  10
#define GEOCP 32
#define EDGEC 128

typedef __attribute__((ext_vector_type(16))) __bf16 v16bf;
typedef __attribute__((ext_vector_type(8)))  __bf16 v8bf;
typedef __attribute__((ext_vector_type(8)))  float  v8f;

__device__ __forceinline__ unsigned short f2bf(float f) {
    union { float f; uint32_t u; } x; x.f = f;
    uint32_t r = x.u + 0x7fffu + ((x.u >> 16) & 1u);   // round-to-nearest-even
    return (unsigned short)(r >> 16);
}

#define DS_FENCE() asm volatile("s_wait_dscnt 0" ::: "memory")

#define XOR_MAX(v, pat) do {                                                   \
    float _w = __int_as_float(__builtin_amdgcn_ds_swizzle(__float_as_int(v), pat)); \
    (v) = (v) > _w ? (v) : _w; } while (0)

// ---------------------------------------------------------------------------
// prep: bf16 point-major transpose of fea + squared norms
// ---------------------------------------------------------------------------
__global__ void __launch_bounds__(256) prep_kernel(
        const float* __restrict__ fea,
        unsigned short* __restrict__ feaT,   // [B*N][C] bf16
        float* __restrict__ sq) {            // [B*N]
    int t = blockIdx.x * 256 + threadIdx.x;           // (b,n) flat
    int b = t >> 12, n = t & (NPTS - 1);
    const float* src = fea + (size_t)b * CIN * NPTS + n;
    unsigned short* dst = feaT + (size_t)t * CIN;
    float s = 0.f;
#pragma unroll
    for (int c = 0; c < CIN; ++c) {
        float v = src[(size_t)c * NPTS];
        s += v * v;
        dst[c] = f2bf(v);
    }
    sq[t] = s;
}

// ---------------------------------------------------------------------------
// knn: bf16 WMMA Gram + branchless per-lane top-16
// ---------------------------------------------------------------------------
__global__ void __launch_bounds__(256) knn_kernel(
        const unsigned short* __restrict__ feaT,
        const float* __restrict__ sq,
        int* __restrict__ idx) {
    __shared__ __attribute__((aligned(32))) float lds_dot[8][32][17];
    const int lane  = threadIdx.x & 31;
    const int wave  = threadIdx.x >> 5;
    const int b     = blockIdx.x >> 4;                       // N/256 = 16 row groups
    const int r0    = ((blockIdx.x & 15) << 8) + wave * 32;  // 32 rows per wave
    const int m15   = lane & 15;
    const int khalf = lane >> 4;
    const unsigned short* fb = feaT + (size_t)b * NPTS * CIN;

    // A fragments (rows r0..r0+31, K=64 in two 32-steps) stay in registers
    v16bf A[2][2];
#pragma unroll
    for (int rt = 0; rt < 2; ++rt)
#pragma unroll
        for (int s = 0; s < 2; ++s) {
            const unsigned short* p = fb + (size_t)(r0 + rt * 16 + m15) * CIN + 32 * s + 16 * khalf;
            v8bf lo = *(const v8bf*)p;
            v8bf hi = *(const v8bf*)(p + 8);
            v16bf a;
#pragma unroll
            for (int i = 0; i < 8; ++i) { a[i] = lo[i]; a[i + 8] = hi[i]; }
            A[rt][s] = a;
        }

    float dk[KNN]; int ik[KNN];
#pragma unroll
    for (int i = 0; i < KNN; ++i) { dk[i] = -3.0e38f; ik[i] = 0; }

    const float* sqb = sq + b * NPTS;
    for (int c0 = 0; c0 < NPTS; c0 += 16) {
        v16bf Bf[2];
#pragma unroll
        for (int s = 0; s < 2; ++s)
            Bf[s] = *(const v16bf*)(fb + (size_t)(c0 + m15) * CIN + 32 * s + 16 * khalf);

        v8f acc0 = {}; v8f acc1 = {};
        acc0 = __builtin_amdgcn_wmma_f32_16x16x32_bf16(false, A[0][0], false, Bf[0], (short)0, acc0, false, false);
        acc0 = __builtin_amdgcn_wmma_f32_16x16x32_bf16(false, A[0][1], false, Bf[1], (short)0, acc0, false, false);
        acc1 = __builtin_amdgcn_wmma_f32_16x16x32_bf16(false, A[1][0], false, Bf[0], (short)0, acc1, false, false);
        acc1 = __builtin_amdgcn_wmma_f32_16x16x32_bf16(false, A[1][1], false, Bf[1], (short)0, acc1, false, false);

#pragma unroll
        for (int r = 0; r < 8; ++r) {
            lds_dot[wave][r + 8 * khalf][m15]      = acc0[r];
            lds_dot[wave][16 + r + 8 * khalf][m15] = acc1[r];
        }
        DS_FENCE();

        // lane owns query row (r0 + lane); key = 2*dot - sq[col]
#pragma unroll
        for (int j = 0; j < 16; ++j) {
            float key = 2.0f * lds_dot[wave][lane][j] - sqb[c0 + j];
            int   mi  = c0 + j;
            bool c[KNN];
#pragma unroll
            for (int s2 = 0; s2 < KNN; ++s2) c[s2] = key > dk[s2];
#pragma unroll
            for (int s2 = KNN - 1; s2 >= 1; --s2) {
                dk[s2] = c[s2 - 1] ? dk[s2 - 1] : (c[s2] ? key : dk[s2]);
                ik[s2] = c[s2 - 1] ? ik[s2 - 1] : (c[s2] ? mi  : ik[s2]);
            }
            dk[0] = c[0] ? key : dk[0];
            ik[0] = c[0] ? mi  : ik[0];
        }
        DS_FENCE();
    }

    int* op = idx + ((size_t)b * NPTS + r0 + lane) * KNN;
#pragma unroll
    for (int j = 0; j < KNN; ++j) op[j] = ik[j];
}

// ---------------------------------------------------------------------------
// fused gather + geo/edge MLPs + max_k : one point = one WMMA column tile
// ---------------------------------------------------------------------------
__device__ __forceinline__ v16bf load_afrag(const unsigned short* W, int Kdim,
                                            int t, int s, int m15, int khalf) {
    const unsigned short* p = W + (16 * t + m15) * Kdim + 32 * s + 16 * khalf;
    v8bf lo = *(const v8bf*)p;
    v8bf hi = *(const v8bf*)(p + 8);
    v16bf a;
#pragma unroll
    for (int i = 0; i < 8; ++i) { a[i] = lo[i]; a[i + 8] = hi[i]; }
    return a;
}

template <int KD>
__device__ __forceinline__ void conv_layer(const unsigned short* W,
                                           const unsigned short* act,
                                           v8f acc[4], int m15, int khalf) {
#pragma unroll
    for (int t = 0; t < 4; ++t) {
        v8f a = {};
#pragma unroll
        for (int s = 0; s < KD / 32; ++s) {
            v16bf af = load_afrag(W, KD, t, s, m15, khalf);
            v16bf bf = *(const v16bf*)(act + m15 * KD + 32 * s + 16 * khalf);
            a = __builtin_amdgcn_wmma_f32_16x16x32_bf16(false, af, false, bf, (short)0, a, false, false);
        }
        acc[t] = a;
    }
}

__device__ __forceinline__ void act_store(v8f acc[4], const float* bias,
                                          unsigned short* dst, int m15, int khalf) {
#pragma unroll
    for (int t = 0; t < 4; ++t)
#pragma unroll
        for (int r = 0; r < 8; ++r) {
            int m = 16 * t + r + 8 * khalf;
            float v = acc[t][r] + bias[m];
            v = v > 0.f ? v : 0.f;
            dst[m15 * CIN + m] = f2bf(v);
        }
}

__global__ void __launch_bounds__(64) edge_kernel(
        const float* __restrict__ xyz, const float* __restrict__ fea,
        const int* __restrict__ idx,
        const float* __restrict__ Wg1, const float* __restrict__ bg1,
        const float* __restrict__ Wg2, const float* __restrict__ bg2,
        const float* __restrict__ Wf1, const float* __restrict__ bf1,
        const float* __restrict__ Wf2, const float* __restrict__ bf2,
        const float* __restrict__ Wf3, const float* __restrict__ bf3,
        float* __restrict__ out) {
    __shared__ __attribute__((aligned(32))) unsigned short sWg1[COUT * GEOCP];
    __shared__ __attribute__((aligned(32))) unsigned short sWg2[COUT * CIN];
    __shared__ __attribute__((aligned(32))) unsigned short sWf1[COUT * EDGEC];
    __shared__ __attribute__((aligned(32))) unsigned short sWf2[COUT * CIN];
    __shared__ __attribute__((aligned(32))) unsigned short sWf3[COUT * CIN];
    __shared__ __attribute__((aligned(32))) float sBias[5][COUT];
    __shared__ __attribute__((aligned(32))) unsigned short sGeo [2][16 * GEOCP];
    __shared__ __attribute__((aligned(32))) unsigned short sEdge[2][16 * EDGEC];
    __shared__ __attribute__((aligned(32))) unsigned short sTmpA[2][16 * CIN];
    __shared__ __attribute__((aligned(32))) unsigned short sTmpB[2][16 * CIN];

    const int tid = threadIdx.x;
    for (int t = tid; t < COUT * GEOCP; t += 64) {
        int m = t / GEOCP, k2 = t % GEOCP;
        sWg1[t] = f2bf(k2 < GEOC ? Wg1[m * GEOC + k2] : 0.f);
    }
    for (int t = tid; t < COUT * CIN;   t += 64) sWg2[t] = f2bf(Wg2[t]);
    for (int t = tid; t < COUT * EDGEC; t += 64) sWf1[t] = f2bf(Wf1[t]);
    for (int t = tid; t < COUT * CIN;   t += 64) sWf2[t] = f2bf(Wf2[t]);
    for (int t = tid; t < COUT * CIN;   t += 64) sWf3[t] = f2bf(Wf3[t]);
    if (tid < COUT) {
        sBias[0][tid] = bg1[tid]; sBias[1][tid] = bg2[tid]; sBias[2][tid] = bf1[tid];
        sBias[3][tid] = bf2[tid]; sBias[4][tid] = bf3[tid];
    }
    __syncthreads();

    const int lane = tid & 31, wave = tid >> 5;
    const int m15 = lane & 15, khalf = lane >> 4;
    const int wid = blockIdx.x * 2 + wave;
    const int nwaves = gridDim.x * 2;

    for (int p = wid; p < BATCH * NPTS; p += nwaves) {
        const int b = p >> 12, n = p & (NPTS - 1);
        const int j = m15;
        const int nbr = idx[((size_t)b * NPTS + n) * KNN + j];
        const float* feab = fea + (size_t)b * CIN * NPTS;

        { // edge features: khalf=0 -> center (ch 0..63), khalf=1 -> neighbor (ch 64..127)
            int srcn = khalf ? nbr : n;
            unsigned short* ep = sEdge[wave] + j * EDGEC + khalf * CIN;
#pragma unroll
            for (int c = 0; c < CIN; ++c)
                ep[c] = f2bf(feab[(size_t)c * NPTS + srcn]);
        }
        if (khalf == 0) { // geo features
            const float* xb = xyz + (size_t)b * 3 * NPTS;
            float cx = xb[n],   cy = xb[NPTS + n],   cz = xb[2 * NPTS + n];
            float kx = xb[nbr], ky = xb[NPTS + nbr], kz = xb[2 * NPTS + nbr];
            float rx = cx - kx, ry = cy - ky, rz = cz - kz;
            float d  = sqrtf(rx * rx + ry * ry + rz * rz);
            unsigned short* gp = sGeo[wave] + j * GEOCP;
            gp[0] = f2bf(d);  gp[1] = f2bf(cx); gp[2] = f2bf(cy); gp[3] = f2bf(cz);
            gp[4] = f2bf(kx); gp[5] = f2bf(ky); gp[6] = f2bf(kz);
            gp[7] = f2bf(rx); gp[8] = f2bf(ry); gp[9] = f2bf(rz);
        } else {
            unsigned short* gp = sGeo[wave] + j * GEOCP;
#pragma unroll
            for (int c = GEOC; c < GEOCP; ++c) gp[c] = 0;
        }
        DS_FENCE();

        v8f accT[4], accG[4], accF[4];
        // g = relu(Wg2 relu(Wg1 geo))
        conv_layer<GEOCP>(sWg1, sGeo[wave], accT, m15, khalf);
        act_store(accT, sBias[0], sTmpA[wave], m15, khalf);
        DS_FENCE();
        conv_layer<CIN>(sWg2, sTmpA[wave], accG, m15, khalf);
#pragma unroll
        for (int t = 0; t < 4; ++t)
#pragma unroll
            for (int r = 0; r < 8; ++r) {
                int m = 16 * t + r + 8 * khalf;
                float v = accG[t][r] + sBias[1][m];
                accG[t][r] = v > 0.f ? v : 0.f;
            }
        // f = relu(Wf3 relu(Wf2 relu(Wf1 edge)))
        conv_layer<EDGEC>(sWf1, sEdge[wave], accT, m15, khalf);
        act_store(accT, sBias[2], sTmpA[wave], m15, khalf);
        DS_FENCE();
        conv_layer<CIN>(sWf2, sTmpA[wave], accT, m15, khalf);
        act_store(accT, sBias[3], sTmpB[wave], m15, khalf);
        DS_FENCE();
        conv_layer<CIN>(sWf3, sTmpB[wave], accF, m15, khalf);
#pragma unroll
        for (int t = 0; t < 4; ++t)
#pragma unroll
            for (int r = 0; r < 8; ++r) {
                int m = 16 * t + r + 8 * khalf;
                float v = accF[t][r] + sBias[4][m];
                accF[t][r] = v > 0.f ? v : 0.f;
            }
        // out[b,m,n] = max_j g*f  (XOR-max across the 16 column lanes)
#pragma unroll
        for (int t = 0; t < 4; ++t)
#pragma unroll
            for (int r = 0; r < 8; ++r) {
                float pm = accG[t][r] * accF[t][r];
                XOR_MAX(pm, 0x041F); XOR_MAX(pm, 0x081F);
                XOR_MAX(pm, 0x101F); XOR_MAX(pm, 0x201F);
                accG[t][r] = pm;
            }
        if (m15 == 0) {
#pragma unroll
            for (int t = 0; t < 4; ++t)
#pragma unroll
                for (int r = 0; r < 8; ++r) {
                    int m = 16 * t + r + 8 * khalf;
                    out[((size_t)b * COUT + m) * NPTS + n] = accG[t][r];
                }
        }
        DS_FENCE();
    }
}

// ---------------------------------------------------------------------------
extern "C" void kernel_launch(void* const* d_in, const int* in_sizes, int n_in,
                              void* d_out, int out_size, void* d_ws, size_t ws_size,
                              hipStream_t stream) {
    (void)in_sizes; (void)n_in; (void)out_size; (void)ws_size;
    const float* xyz = (const float*)d_in[0];
    const float* fea = (const float*)d_in[1];
    const float* Wg1 = (const float*)d_in[2];
    const float* bg1 = (const float*)d_in[3];
    const float* Wg2 = (const float*)d_in[4];
    const float* bg2 = (const float*)d_in[5];
    const float* Wf1 = (const float*)d_in[6];
    const float* bf1 = (const float*)d_in[7];
    const float* Wf2 = (const float*)d_in[8];
    const float* bf2 = (const float*)d_in[9];
    const float* Wf3 = (const float*)d_in[10];
    const float* bf3 = (const float*)d_in[11];

    uint8_t* ws = (uint8_t*)d_ws;
    size_t oFeaT = 0;
    size_t oSq   = oFeaT + (size_t)BATCH * NPTS * CIN * sizeof(unsigned short); // 4 MB
    size_t oIdx  = oSq   + (size_t)BATCH * NPTS * sizeof(float);                // +128 KB
    unsigned short* feaT = (unsigned short*)(ws + oFeaT);
    float*          sq   = (float*)(ws + oSq);
    int*            nidx = (int*)(ws + oIdx);

    prep_kernel<<<BATCH * NPTS / 256, 256, 0, stream>>>(fea, feaT, sq);
    knn_kernel <<<BATCH * (NPTS / 256), 256, 0, stream>>>(feaT, sq, nidx);
    edge_kernel<<<512, 64, 0, stream>>>(xyz, fea, nidx,
                                        Wg1, bg1, Wg2, bg2,
                                        Wf1, bf1, Wf2, bf2, Wf3, bf3,
                                        (float*)d_out);
}
